// TinyAttentionBlock_46943992545481
// MI455X (gfx1250) — matrix operand order repaired and checked by reading, not compiled, not run on hardware
//
#include <hip/hip_runtime.h>

// TinyAttentionBlock on gfx1250 (MI455X):
//   LN1 -> QKV GEMM -> attention (QK^T + rel bias, softmax, PV) -> proj -> LN2
// All GEMMs use V_WMMA_F32_16X16X32_BF16 with bf16x3 fp32 emulation:
//   x = hi + lo (bf16 split);  A*B ~= Ah*Bh + Ah*Bl + Al*Bh  (fp32 accumulate)
// => fp32-class accuracy at 3 WMMAs per K=32 (vs 8 for the f32 K=4 WMMA).

#define SEQ_LEN 2048
#define EMBED   256
#define BATCH   8
#define ROWS    (BATCH * SEQ_LEN)   // 16384

typedef float          v2f   __attribute__((ext_vector_type(2)));
typedef float          v8f   __attribute__((ext_vector_type(8)));
typedef __bf16         v16bf __attribute__((ext_vector_type(16)));
typedef unsigned short v8us  __attribute__((ext_vector_type(8)));
typedef unsigned short u16;

__device__ __forceinline__ u16 f32_to_bf16(float f) {
    unsigned int u = __float_as_uint(f);
    u += 0x7FFFu + ((u >> 16) & 1u);          // round-to-nearest-even
    return (u16)(u >> 16);
}
__device__ __forceinline__ float bf16_to_f32(u16 h) {
    return __uint_as_float(((unsigned int)h) << 16);
}
__device__ __forceinline__ void bf16_split(float x, u16& h, u16& l) {
    h = f32_to_bf16(x);
    l = f32_to_bf16(x - bf16_to_f32(h));
}

union U16x16 { v8us u[2]; u16 s[16]; v16bf v; };

// A operand (16x32 bf16): lane holds K = {c+8*hi .. +7} and {c+16+8*hi .. +7}
__device__ __forceinline__ v16bf load_a16(const u16* p, int hi) {
    U16x16 r;
    r.u[0] = *(const v8us*)(p + 8 * hi);
    r.u[1] = *(const v8us*)(p + 16 + 8 * hi);
    return r.v;
}
// B operand (32x16 bf16): lane holds K = {c+8*hi .. +7} and {c+16+8*hi .. +7} of row n=lane&15
// (rows taken from a pre-transposed matrix so this is contiguous)
__device__ __forceinline__ v16bf load_b16(const u16* p, int hi) {
    U16x16 r;
    r.u[0] = *(const v8us*)(p + 8 * hi);
    r.u[1] = *(const v8us*)(p + 16 + 8 * hi);
    return r.v;
}

__device__ __forceinline__ v8f wmma_bf16(v16bf a, v16bf b, v8f c) {
    return __builtin_amdgcn_wmma_f32_16x16x32_bf16(
        false, a, false, b, (short)0, c, false, false);
}

// ---------------------------------------------------------------------------
// LayerNorm over rows of length 256. One block (256 threads) per row.
// ---------------------------------------------------------------------------
__global__ __launch_bounds__(256)
void ln_kernel(const float* __restrict__ x, const float* __restrict__ g,
               const float* __restrict__ b, float* __restrict__ y) {
    __shared__ float sred[256];
    const int row = blockIdx.x;
    const int t   = threadIdx.x;
    const float v = x[(size_t)row * EMBED + t];

    sred[t] = v;
    __syncthreads();
    for (int s = 128; s > 0; s >>= 1) {
        if (t < s) sred[t] += sred[t + s];
        __syncthreads();
    }
    const float mu = sred[0] * (1.0f / EMBED);
    __syncthreads();
    const float d = v - mu;
    sred[t] = d * d;
    __syncthreads();
    for (int s = 128; s > 0; s >>= 1) {
        if (t < s) sred[t] += sred[t + s];
        __syncthreads();
    }
    const float var = sred[0] * (1.0f / EMBED);
    y[(size_t)row * EMBED + t] = d * rsqrtf(var + 1e-5f) * g[t] + b[t];
}

// ---------------------------------------------------------------------------
// Elementwise fp32 -> (bf16 hi, bf16 lo) split.
// ---------------------------------------------------------------------------
__global__ __launch_bounds__(256)
void split_kernel(const float* __restrict__ x, u16* __restrict__ h,
                  u16* __restrict__ l) {
    const size_t i = (size_t)blockIdx.x * 256 + threadIdx.x;
    bf16_split(x[i], h[i], l[i]);
}

// W[K,Nout] -> transposed split WT_hi/lo [Nout,K]
__global__ __launch_bounds__(256)
void wsplitT_kernel(const float* __restrict__ W, u16* __restrict__ th,
                    u16* __restrict__ tl, int K, int Nout) {
    const size_t i = (size_t)blockIdx.x * 256 + threadIdx.x;
    const int k = (int)(i / Nout), n = (int)(i - (size_t)k * Nout);
    u16 h, l;
    bf16_split(W[i], h, l);
    th[(size_t)n * K + k] = h;
    tl[(size_t)n * K + k] = l;
}

// qkv[B,N,768] -> q_hi/lo, k_hi/lo row-major [B*N,256]; vT_hi/lo [B,256,2048]
__global__ __launch_bounds__(256)
void qkv_split_kernel(const float* __restrict__ qkv,
                      u16* __restrict__ qh, u16* __restrict__ ql,
                      u16* __restrict__ kh, u16* __restrict__ kl,
                      u16* __restrict__ vth, u16* __restrict__ vtl) {
    const size_t i = (size_t)blockIdx.x * 256 + threadIdx.x;   // over B*N*C
    const int c = (int)(i & 255);
    const int n = (int)((i >> 8) & (SEQ_LEN - 1));
    const int b = (int)(i >> 19);
    const float* src = qkv + ((size_t)b * SEQ_LEN + n) * 768;
    const size_t ro = ((size_t)b * SEQ_LEN + n) * EMBED + c;
    const size_t to = ((size_t)b * EMBED + c) * SEQ_LEN + n;
    bf16_split(src[c],       qh[ro],  ql[ro]);
    bf16_split(src[256 + c], kh[ro],  kl[ro]);
    bf16_split(src[512 + c], vth[to], vtl[to]);
}

// ---------------------------------------------------------------------------
// GEMM: out[ROWS,Nout] = A[ROWS,256] @ W[256,Nout] + bias, bf16x3 WMMA.
// A given as hi/lo row-major; W given as hi/lo TRANSPOSED [Nout,256].
// One 16x16 tile per wave, 8 waves/block; K-loop in steps of 32.
// ---------------------------------------------------------------------------
__global__ __launch_bounds__(256)
void gemm_bf16x3_kernel(const u16* __restrict__ Ah, const u16* __restrict__ Al,
                        const u16* __restrict__ WTh, const u16* __restrict__ WTl,
                        const float* __restrict__ bias, float* __restrict__ out,
                        int Nout) {
    const int wave = threadIdx.x >> 5, lane = threadIdx.x & 31;
    const int lh = lane & 15, hi = lane >> 4;
    const int ntile_n = Nout >> 4;
    const int tile = blockIdx.x * 8 + wave;
    const int mt = tile / ntile_n, nt = tile - mt * ntile_n;
    const int m0 = mt << 4, n0 = nt << 4;

    const u16* arh = Ah  + (size_t)(m0 + lh) * EMBED;
    const u16* arl = Al  + (size_t)(m0 + lh) * EMBED;
    const u16* brh = WTh + (size_t)(n0 + lh) * EMBED;
    const u16* brl = WTl + (size_t)(n0 + lh) * EMBED;

    v8f acc = {};
#pragma unroll
    for (int c = 0; c < EMBED; c += 32) {
        v16bf ah = load_a16(arh + c, hi);
        v16bf al = load_a16(arl + c, hi);
        v16bf bh = load_b16(brh + c, hi);
        v16bf bl = load_b16(brl + c, hi);
        acc = wmma_bf16(ah, bh, acc);
        acc = wmma_bf16(ah, bl, acc);
        acc = wmma_bf16(al, bh, acc);
    }

    const float bb = bias[n0 + lh];
    float* orow = out + n0 + lh;
#pragma unroll
    for (int r = 0; r < 8; ++r)
        orow[(size_t)(m0 + r + 8 * hi) * Nout] = acc[r] + bb;
}

// ---------------------------------------------------------------------------
// Fused attention for one (batch, 16-query tile):
//   S[16,2048] = Q @ K^T + rel_bias  (bf16x3 WMMA, S kept fp32 in LDS)
//   block softmax; scale/rowsum folded into the output store
//   O[16,256] = P @ V                (bf16x3 WMMA; P split on the fly,
//                                     V pre-split & transposed [C,N])
// LDS: 128K scores + 2*8K Q(hi/lo) + 1K red  (~145 KB < 320 KB/WG)
// ---------------------------------------------------------------------------
__global__ __launch_bounds__(256)
void attn_kernel(const u16* __restrict__ qh, const u16* __restrict__ ql,
                 const u16* __restrict__ kh, const u16* __restrict__ kl,
                 const u16* __restrict__ vth, const u16* __restrict__ vtl,
                 const float* __restrict__ table,
                 const float* __restrict__ scale_p, float* __restrict__ o) {
    __shared__ float sS[16 * SEQ_LEN];
    __shared__ u16   sQh[16 * EMBED];
    __shared__ u16   sQl[16 * EMBED];
    __shared__ float red[256];
    __shared__ float rs[16];

    const int b   = blockIdx.y;
    const int qi0 = blockIdx.x << 4;
    const int t = threadIdx.x, wave = t >> 5, lane = t & 31;
    const int lh = lane & 15, hi = lane >> 4;

    // Stage the 16x256 Q tile (hi/lo) into LDS.
    for (int i = t; i < 16 * EMBED; i += 256) {
        const size_t src = ((size_t)b * SEQ_LEN + qi0 + (i >> 8)) * EMBED + (i & 255);
        sQh[i] = qh[src];
        sQl[i] = ql[src];
    }
    __syncthreads();

    // ---- QK^T: each wave owns every 8th key tile. --------------------------
    const u16* qrh = sQh + lh * EMBED;
    const u16* qrl = sQl + lh * EMBED;
    for (int jt = wave; jt < SEQ_LEN / 16; jt += 8) {
        const size_t krow = ((size_t)b * SEQ_LEN + (jt << 4) + lh) * EMBED;
        const u16* krh = kh + krow;
        const u16* krl = kl + krow;
        __builtin_prefetch(kh + krow + (size_t)8 * 16 * EMBED, 0, 1); // next tile
        v8f acc = {};
#pragma unroll
        for (int c = 0; c < EMBED; c += 32) {
            v16bf ah = load_a16(qrh + c, hi);
            v16bf al = load_a16(qrl + c, hi);
            v16bf bh = load_b16(krh + c, hi);
            v16bf bl = load_b16(krl + c, hi);
            acc = wmma_bf16(ah, bh, acc);
            acc = wmma_bf16(ah, bl, acc);
            acc = wmma_bf16(al, bh, acc);
        }
#pragma unroll
        for (int r = 0; r < 8; ++r) {
            const int M = r + 8 * hi;
            const int rel = (jt << 4) + lh - (qi0 + M) + (SEQ_LEN - 1);
            sS[M * SEQ_LEN + (jt << 4) + lh] = acc[r] + table[rel];
        }
    }
    __syncthreads();

    // ---- softmax: 16 threads per row, 128 keys per thread. -----------------
    {
        const int row = t >> 4, sub = t & 15;
        float* srow = sS + row * SEQ_LEN + sub * 128;
        float lmax = -3.4e38f;
        for (int i = 0; i < 128; ++i) lmax = fmaxf(lmax, srow[i]);
        red[t] = lmax;
        __syncthreads();
        float rowmax = -3.4e38f;
#pragma unroll
        for (int k = 0; k < 16; ++k) rowmax = fmaxf(rowmax, red[(row << 4) + k]);
        __syncthreads();
        float lsum = 0.0f;
        for (int i = 0; i < 128; ++i) {
            const float e = __expf(srow[i] - rowmax);
            srow[i] = e;
            lsum += e;
        }
        red[t] = lsum;
        __syncthreads();
        float rowsum = 0.0f;
#pragma unroll
        for (int k = 0; k < 16; ++k) rowsum += red[(row << 4) + k];
        if (sub == 0) rs[row] = scale_p[0] / rowsum;   // post-softmax scale
    }
    __syncthreads();

    // ---- P @ V: each wave owns 2 of the 16 output column tiles. ------------
    const float* prow = sS + lh * SEQ_LEN;
    for (int ct = wave * 2; ct < wave * 2 + 2; ++ct) {
        const int c0 = ct << 4;
        const size_t vrow = ((size_t)b * EMBED + c0 + lh) * SEQ_LEN;
        const u16* vrh = vth + vrow;
        const u16* vrl = vtl + vrow;
        v8f acc = {};
        for (int kb = 0; kb < SEQ_LEN; kb += 32) {
            __builtin_prefetch(vrh + kb + 512, 0, 1);
            // split the fp32 probabilities from LDS on the fly (A layout)
            U16x16 ah, al;
            const float* p0 = prow + kb + 8 * hi;
            const float* p1 = prow + kb + 16 + 8 * hi;
#pragma unroll
            for (int j = 0; j < 8; ++j) {
                bf16_split(p0[j], ah.s[j], al.s[j]);
                bf16_split(p1[j], ah.s[8 + j], al.s[8 + j]);
            }
            v16bf bh = load_b16(vrh + kb, hi);
            v16bf bl = load_b16(vrl + kb, hi);
            acc = wmma_bf16(ah.v, bh, acc);
            acc = wmma_bf16(ah.v, bl, acc);
            acc = wmma_bf16(al.v, bh, acc);
        }
#pragma unroll
        for (int r = 0; r < 8; ++r) {
            const int M = r + 8 * hi;
            o[((size_t)b * SEQ_LEN + qi0 + M) * EMBED + c0 + lh] = acc[r] * rs[M];
        }
    }
}

// ---------------------------------------------------------------------------
extern "C" void kernel_launch(void* const* d_in, const int* in_sizes, int n_in,
                              void* d_out, int out_size, void* d_ws, size_t ws_size,
                              hipStream_t stream) {
    const float* x          = (const float*)d_in[0];
    const float* ln1_g      = (const float*)d_in[1];
    const float* ln1_b      = (const float*)d_in[2];
    const float* qkv_w      = (const float*)d_in[3];
    const float* qkv_b      = (const float*)d_in[4];
    const float* bias_table = (const float*)d_in[5];
    const float* scale      = (const float*)d_in[6];
    const float* proj_w     = (const float*)d_in[7];
    const float* proj_b     = (const float*)d_in[8];
    const float* ln2_g      = (const float*)d_in[9];
    const float* ln2_b      = (const float*)d_in[10];

    const size_t nBNC = (size_t)ROWS * EMBED;            // 4,194,304
    char* p = (char*)d_ws;
    float* qkvbuf = (float*)p;  p += (size_t)ROWS * 768 * 4;   // 50.3 MB (reused as proj tmp)
    float* hbuf   = (float*)p;  p += nBNC * 4;                 // 16.8 MB (reused as O)
    u16* hh  = (u16*)p; p += nBNC * 2;                         // also O_hi
    u16* hl  = (u16*)p; p += nBNC * 2;                         // also O_lo
    u16* qhh = (u16*)p; p += nBNC * 2;
    u16* qll = (u16*)p; p += nBNC * 2;
    u16* khh = (u16*)p; p += nBNC * 2;
    u16* kll = (u16*)p; p += nBNC * 2;
    u16* vth = (u16*)p; p += nBNC * 2;
    u16* vtl = (u16*)p; p += nBNC * 2;
    u16* wqh = (u16*)p; p += (size_t)256 * 768 * 2;
    u16* wql = (u16*)p; p += (size_t)256 * 768 * 2;
    u16* wph = (u16*)p; p += (size_t)256 * 256 * 2;
    u16* wpl = (u16*)p; p += (size_t)256 * 256 * 2;
    float* tmpbuf = qkvbuf;                                    // q/k/v fp32 dead by then

    // 1) LN1
    ln_kernel<<<ROWS, 256, 0, stream>>>(x, ln1_g, ln1_b, hbuf);
    // 2) bf16 hi/lo splits of h and qkv_w^T
    split_kernel<<<(int)(nBNC / 256), 256, 0, stream>>>(hbuf, hh, hl);
    wsplitT_kernel<<<(256 * 768) / 256, 256, 0, stream>>>(qkv_w, wqh, wql, 256, 768);
    // 3) QKV GEMM (bf16x3): [16384,256]x[256,768] -> 49152 tiles / 8 waves
    gemm_bf16x3_kernel<<<(ROWS / 16) * (768 / 16) / 8, 256, 0, stream>>>(
        hh, hl, wqh, wql, qkv_b, qkvbuf, 768);
    // 4) split q/k row-major, v transposed
    qkv_split_kernel<<<(int)(nBNC / 256), 256, 0, stream>>>(
        qkvbuf, qhh, qll, khh, kll, vth, vtl);
    // 5) fused attention -> O (over hbuf)
    attn_kernel<<<dim3(SEQ_LEN / 16, BATCH), 256, 0, stream>>>(
        qhh, qll, khh, kll, vth, vtl, bias_table, scale, hbuf);
    // 6) split O, transpose-split proj_w
    split_kernel<<<(int)(nBNC / 256), 256, 0, stream>>>(hbuf, hh, hl);
    wsplitT_kernel<<<(256 * 256) / 256, 256, 0, stream>>>(proj_w, wph, wpl, 256, 256);
    // 7) proj GEMM (bf16x3) -> tmp (over qkv region)
    gemm_bf16x3_kernel<<<(ROWS / 16) * (256 / 16) / 8, 256, 0, stream>>>(
        hh, hl, wph, wpl, proj_b, tmpbuf, 256);
    // 8) LN2 -> final output
    ln_kernel<<<ROWS, 256, 0, stream>>>(tmpbuf, ln2_g, ln2_b, (float*)d_out);
}
